// AAEncoder_9680856285424
// MI455X (gfx1250) — compile-verified, matrix operand order checked by reading, not run on hardware
//
#include <hip/hip_runtime.h>
#include <hip/hip_bf16.h>
#include <stdint.h>

// ============================================================================
// AAEncoder for MI455X (gfx1250, wave32, WMMA).
// Fused per-edge pipeline: all 6 edge-side 128x128 GEMMs run in one kernel on
// 16-edge tiles held in LDS (avoids ~1.2 GB of materialized [E,128] traffic).
// Matrix math: v_wmma_f32_16x16x32_f16 (f16 A/B, f32 accumulate).
// ce_n[dst] gather: GLOBAL_LOAD_ASYNC_TO_LDS_B128 (ASYNCcnt), overlapped with
// the neighbor-embedding GEMMs; segment-sum via global_atomic_add_f32.
// ============================================================================

typedef __attribute__((ext_vector_type(16))) _Float16 v16h;
typedef __attribute__((ext_vector_type(8)))  float    v8f;

#define LDA 132          // padded LDS row stride (floats)
#define LN_EPS 1e-5f

// ---------------------------------------------------------------------------
// Input index map. Assumes the harness flattens setup_inputs() with JAX tree
// semantics (dict keys sorted alphabetically at every nesting level):
// bos_mask, edge_attr, edge_index, nodes, params{...}, rotate_mat, t
// ---------------------------------------------------------------------------
#define I_BOS_MASK   0
#define I_EDGE_ATTR  1
#define I_EDGE_INDEX 2
#define I_NODES      3
#define I_BOS_TOKEN  4
#define I_C_L1_B     5
#define I_C_L1_W     6
#define I_C_L2_B     7
#define I_C_L2_W     8
#define I_C_L3_B     9
#define I_C_L3_W    10
#define I_C_LN1_B   11
#define I_C_LN1_G   12
#define I_C_LN2_B   13
#define I_C_LN2_G   14
#define I_C_LN3_B   15
#define I_C_LN3_G   16
#define I_HH_B      17
#define I_HH_W      18
#define I_IH_B      19
#define I_IH_W      20
#define I_K_B       21
#define I_K_W       22
#define I_Q_B       23
#define I_Q_W       24
#define I_SELF_B    25
#define I_SELF_W    26
#define I_V_B       27
#define I_V_W       28
#define I_MLP1_B    29
#define I_MLP1_W    30
#define I_MLP2_B    31
#define I_MLP2_W    32
#define I_AG_L_B    33
#define I_AG_L_W    34
#define I_AG_LN1_B  35
#define I_AG_LN1_G  36
#define I_AG_LN2_B  37
#define I_AG_LN2_G  38
#define I_NE_L1_B   39
#define I_NE_L1_W   40
#define I_NE_L2_B   41
#define I_NE_L2_W   42
#define I_NE_LN_B   43
#define I_NE_LN_G   44
#define I_NN_L1_B   45
#define I_NN_L1_W   46
#define I_NN_L2_B   47
#define I_NN_L2_W   48
#define I_NN_LN_B   49
#define I_NN_LN_G   50
#define I_NORM1_B   51
#define I_NORM1_G   52
#define I_NORM2_B   53
#define I_NORM2_G   54
#define I_OUTP_B    55
#define I_OUTP_W    56
#define I_ROT       57
#define I_T         58

// ---------------------------------------------------------------------------
// WMMA fragment helpers (CDNA5 ISA 7.12.2 layouts, wave32)
// f16 A 16x32: lanes 0-15 row M, K 0..7 & 16..23; lanes 16-31 K 8..15 & 24..31
// ---------------------------------------------------------------------------
__device__ __host__ __forceinline__ int kmap(int e, int hi) {
  int i = e >> 1, s = e & 1;
  int base = (i < 4) ? (2 * i + s) : (16 + 2 * (i - 4) + s);
  return base + 8 * hi;
}

__device__ __forceinline__ v16h load_a_frag(const float* A, int kk, int lane) {
  int m = lane & 15, hi = lane >> 4;
  v16h a;
#pragma unroll
  for (int e = 0; e < 16; ++e)
    a[e] = (_Float16)A[m * LDA + kk * 32 + kmap(e, hi)];
  return a;
}

// B fragments pre-packed: ((kt*NNT + nt)*32 + lane)*16 + e  (contiguous/lane)
__device__ __forceinline__ v16h load_b_frag(const _Float16* wf, int NNT,
                                            int kt, int nt, int lane) {
  return *(const v16h*)(wf + ((((size_t)kt * NNT + nt) * 32 + lane) << 4));
}

// D = A[16 x 32*ktn] * B + (bias / +=O), D layout: lane n = lane&15,
// VGPR r -> M = r + 8*(lane>>4)
__device__ void gemm_frag(const float* A, const _Float16* wf, int NNT,
                          int kt0, int ktn, int nt0, int ntn,
                          float* O, const float* bias, bool accum, bool relu,
                          int lane) {
  __syncthreads();
  __builtin_prefetch(wf + (((size_t)kt0 * NNT + nt0) << 9), 0, 3);
  int nlane = lane & 15, mb = (lane >> 4) * 8;
  for (int t = 0; t < ntn; ++t) {
    v8f c = {};
    for (int kk = 0; kk < ktn; ++kk) {
      v16h av = load_a_frag(A, kk, lane);
      v16h bv = load_b_frag(wf, NNT, kt0 + kk, nt0 + t, lane);
      c = __builtin_amdgcn_wmma_f32_16x16x32_f16(false, av, false, bv,
                                                 (short)0, c, false, false);
    }
    int nl = t * 16 + nlane;
    float bb = bias ? bias[nl] : 0.f;
#pragma unroll
    for (int r = 0; r < 8; ++r) {
      float val = c[r] + bb;
      float* p = O + (mb + r) * LDA + nl;
      if (accum) val += *p;
      if (relu)  val = fmaxf(val, 0.f);
      *p = val;
    }
  }
  __syncthreads();
}

// Row-wise LayerNorm over 128 cols of a 16-row LDS tile, optional ReLU.
__device__ void lnorm(float* buf, const float* g, const float* b, bool relu,
                      int lane, float* red) {
  int row = lane & 15, c0 = (lane >> 4) * 64;
  __syncthreads();
  float s = 0.f, s2 = 0.f;
  for (int c = c0; c < c0 + 64; ++c) {
    float x = buf[row * LDA + c]; s += x; s2 += x * x;
  }
  red[lane] = s; red[32 + lane] = s2;
  __syncthreads();
  float S  = red[row] + red[row + 16];
  float S2 = red[32 + row] + red[32 + row + 16];
  float mean = S * (1.f / 128.f);
  float var  = S2 * (1.f / 128.f) - mean * mean;
  float inv  = rsqrtf(var + LN_EPS);
  for (int c = c0; c < c0 + 64; ++c) {
    float x = (buf[row * LDA + c] - mean) * inv * g[c] + b[c];
    if (relu) x = fmaxf(x, 0.f);
    buf[row * LDA + c] = x;
  }
  __syncthreads();
}

// ---------------------------------------------------------------------------
// Weight repack: fp32 [dout,din] row-major -> f16 WMMA B-fragment layout
// (B[k][n] = W[n][k]); one thread per element.
// ---------------------------------------------------------------------------
__global__ void k_prep_weight(const float* __restrict__ w,
                              _Float16* __restrict__ out, int dout, int din) {
  int idx = blockIdx.x * blockDim.x + threadIdx.x;
  if (idx >= dout * din) return;
  int NNT  = dout >> 4;
  int e    = idx & 15;
  int lane = (idx >> 4) & 31;
  int rest = idx >> 9;
  int nt   = rest % NNT;
  int kt   = rest / NNT;
  int n = nt * 16 + (lane & 15);
  int k = kt * 32 + kmap(e, lane >> 4);
  out[idx] = (_Float16)w[(size_t)n * din + k];
}

__global__ void k_zero(float* __restrict__ p, int n) {
  int i = blockIdx.x * blockDim.x + threadIdx.x;
  if (i < n) p[i] = 0.f;
}

// ---------------------------------------------------------------------------
// Kernel 1: center embedding + norm1.  One wave / 16 nodes.
// ---------------------------------------------------------------------------
struct NodeArgs {
  const float *nodes, *rot;
  const float *l1w, *l1b, *ln1g, *ln1b;
  const _Float16* l2f; const float *l2b, *ln2g, *ln2b;
  const _Float16* l3f; const float *l3b, *ln3g, *ln3b;
  const float *n1g, *n1b;
  float *ce, *cen;
  int N;
};

__global__ __launch_bounds__(32) void k_node_embed(NodeArgs a) {
  __shared__ float bA[16][LDA], bB[16][LDA];
  __shared__ float red[64];
  int lane = threadIdx.x;
  int row = lane & 15, c0 = (lane >> 4) * 64;
  int node = blockIdx.x * 16 + row;
  int nc = min(node, a.N - 1);

  float x0 = a.nodes[(size_t)nc * 2 + 0], x1 = a.nodes[(size_t)nc * 2 + 1];
  const float* R = a.rot + (size_t)nc * 4;
  float r0 = x0 * R[0] + x1 * R[2];       // einsum ni,nij->nj
  float r1 = x0 * R[1] + x1 * R[3];

  for (int c = c0; c < c0 + 64; ++c)       // l1 (2 -> 128), plain FMAs
    bA[row][c] = a.l1w[c * 2] * r0 + a.l1w[c * 2 + 1] * r1 + a.l1b[c];
  lnorm(&bA[0][0], a.ln1g, a.ln1b, true, lane, red);
  gemm_frag(&bA[0][0], a.l2f, 8, 0, 4, 0, 8, &bB[0][0], a.l2b, false, false, lane);
  lnorm(&bB[0][0], a.ln2g, a.ln2b, true, lane, red);
  gemm_frag(&bB[0][0], a.l3f, 8, 0, 4, 0, 8, &bA[0][0], a.l3b, false, false, lane);
  lnorm(&bA[0][0], a.ln3g, a.ln3b, false, lane, red);   // ce

  if (node < a.N)
    for (int c = c0; c < c0 + 64; ++c) a.ce[(size_t)node * 128 + c] = bA[row][c];
  lnorm(&bA[0][0], a.n1g, a.n1b, false, lane, red);     // ce_n = norm1(ce)
  if (node < a.N)
    for (int c = c0; c < c0 + 64; ++c) a.cen[(size_t)node * 128 + c] = bA[row][c];
}

// ---------------------------------------------------------------------------
// Kernel 2: fused edge pipeline.  One wave / 16 edges.  192 v_wmma / tile.
// ce_n[dst] gather runs asynchronously (ASYNCcnt) under the nbr-embed GEMMs.
// ---------------------------------------------------------------------------
struct EdgeArgs {
  const int* ei;
  const float *nodes, *eattr, *rot, *cen;
  const float *nnl1w, *nnl1b, *nnlng, *nnlnb;
  const _Float16* nnl2f; const float* nnl2b;
  const float *nel1w, *nel1b, *nelng, *nelnb;
  const _Float16* nel2f; const float* nel2b;
  const float *ag1g, *ag1b;
  const _Float16* agf; const float *agb, *ag2g, *ag2b;
  const _Float16* qf; const float* qb;
  const _Float16* kf; const float* kb;
  const _Float16* vf; const float* vb;
  float* msg;
  int E;
};

// one B128 async global->LDS transfer at compile-time byte offset OFF;
// INST_OFFSET is applied to BOTH the LDS and global addresses (ISA 10.x).
#define ASYNC_B128(OFF)                                                        \
  asm volatile("global_load_async_to_lds_b128 %0, %1, off offset:" #OFF        \
               :: "v"(ldst), "v"(gsrc) : "memory")

__global__ __launch_bounds__(32) void k_edge(EdgeArgs a) {
  __shared__ float bA[16][LDA], bB[16][LDA], bC[16][LDA], bG[16][LDA];
  __shared__ float red[64];
  __shared__ float alphaS[16][8];
  __shared__ float xjS[16][2], erS[16][2];
  __shared__ int dstS[16];
  int lane = threadIdx.x;
  int e0 = blockIdx.x * 16;

  if (lane < 16) {                               // gather + rotate per edge
    int e = min(e0 + lane, a.E - 1);
    int s = a.ei[e];
    int d = a.ei[(size_t)a.E + e];
    dstS[lane] = d;
    const float* R = a.rot + (size_t)d * 4;
    float p0 = a.nodes[(size_t)s * 2], p1 = a.nodes[(size_t)s * 2 + 1];
    xjS[lane][0] = p0 * R[0] + p1 * R[2];
    xjS[lane][1] = p0 * R[1] + p1 * R[3];
    float q0 = a.eattr[(size_t)e * 2], q1 = a.eattr[(size_t)e * 2 + 1];
    erS[lane][0] = q0 * R[0] + q1 * R[2];
    erS[lane][1] = q0 * R[1] + q1 * R[3];
  }
  __syncthreads();

  int row = lane & 15, c0 = (lane >> 4) * 64;

  {  // kick off async gather of ce_n[dst] -> bG; each lane copies 64 floats
     // (16 x B128).  Overlaps with the three nbr-embed GEMM+LN stages below.
    const float* gsrc = a.cen + (size_t)dstS[row] * 128 + c0;
    uint32_t ldst = (uint32_t)(uintptr_t)&bG[row][c0];   // wave-relative LDS
    ASYNC_B128(0);   ASYNC_B128(16);  ASYNC_B128(32);  ASYNC_B128(48);
    ASYNC_B128(64);  ASYNC_B128(80);  ASYNC_B128(96);  ASYNC_B128(112);
    ASYNC_B128(128); ASYNC_B128(144); ASYNC_B128(160); ASYNC_B128(176);
    ASYNC_B128(192); ASYNC_B128(208); ASYNC_B128(224); ASYNC_B128(240);
  }

  float x0 = xjS[row][0], x1 = xjS[row][1];
  float y0 = erS[row][0], y1 = erS[row][1];
  for (int c = c0; c < c0 + 64; ++c) {           // the two 2->128 linears
    bA[row][c] = a.nnl1w[c * 2] * x0 + a.nnl1w[c * 2 + 1] * x1 + a.nnl1b[c];
    bB[row][c] = a.nel1w[c * 2] * y0 + a.nel1w[c * 2 + 1] * y1 + a.nel1b[c];
  }
  lnorm(&bA[0][0], a.nnlng, a.nnlnb, true, lane, red);
  lnorm(&bB[0][0], a.nelng, a.nelnb, true, lane, red);
  gemm_frag(&bA[0][0], a.nnl2f, 8, 0, 4, 0, 8, &bC[0][0], a.nnl2b, false, false, lane);
  gemm_frag(&bB[0][0], a.nel2f, 8, 0, 4, 0, 8, &bC[0][0], a.nel2b, true,  false, lane);
  lnorm(&bC[0][0], a.ag1g, a.ag1b, true, lane, red);
  gemm_frag(&bC[0][0], a.agf,  8, 0, 4, 0, 8, &bB[0][0], a.agb, false, false, lane);
  lnorm(&bB[0][0], a.ag2g, a.ag2b, false, lane, red);       // bB = nbr

  asm volatile("s_wait_asynccnt 0x0" ::: "memory");          // gather done
  __syncthreads();

  gemm_frag(&bG[0][0], a.qf, 8, 0, 4, 0, 8, &bC[0][0], a.qb, false, false, lane); // q
  gemm_frag(&bB[0][0], a.kf, 8, 0, 4, 0, 8, &bA[0][0], a.kb, false, false, lane); // k

  {                                              // alpha[e][h] = q.k / sqrt(16)
    int hb = (lane >> 4) * 4;
    for (int hh = 0; hh < 4; ++hh) {
      int hd = hb + hh; float sdot = 0.f;
      for (int dd = 0; dd < 16; ++dd)
        sdot += bC[row][hd * 16 + dd] * bA[row][hd * 16 + dd];
      alphaS[row][hd] = sdot * 0.25f;
    }
  }
  __syncthreads();
  if (lane < 16) {                               // softmax over 8 heads
    float mx = alphaS[lane][0];
    for (int hd = 1; hd < 8; ++hd) mx = fmaxf(mx, alphaS[lane][hd]);
    float ss = 0.f, ex[8];
    for (int hd = 0; hd < 8; ++hd) { ex[hd] = expf(alphaS[lane][hd] - mx); ss += ex[hd]; }
    float is = 1.f / ss;
    for (int hd = 0; hd < 8; ++hd) alphaS[lane][hd] = ex[hd] * is;
  }
  __syncthreads();
  gemm_frag(&bB[0][0], a.vf, 8, 0, 4, 0, 8, &bA[0][0], a.vb, false, false, lane); // v

  if (e0 + row < a.E) {                          // segment_sum via f32 atomics
    float* mrow = a.msg + (size_t)dstS[row] * 128;
    for (int c = c0; c < c0 + 64; ++c)
      atomicAdd(&mrow[c], bA[row][c] * alphaS[row][c >> 4]);
  }
}

// ---------------------------------------------------------------------------
// Kernel 3: out_proj, gating, norm2, MLP (512 hidden, chunked).  16 nodes/wave.
// ---------------------------------------------------------------------------
struct OutArgs {
  const float *msgacc, *ce, *cen;
  const _Float16* opf; const float* opb;
  const _Float16* ihf; const float* ihb;
  const _Float16* hhf; const float* hhb;
  const _Float16* sf;  const float* sb;
  const float *n2g, *n2b;
  const _Float16* m1f; const float* m1b;
  const _Float16* m2f; const float* m2b;
  float* out;
  int N;
};

__global__ __launch_bounds__(32) void k_node_out(OutArgs a) {
  __shared__ float b0[16][LDA], b1[16][LDA], b2[16][LDA], b3[16][LDA];
  __shared__ float red[64];
  int lane = threadIdx.x;
  int row = lane & 15, c0 = (lane >> 4) * 64;
  int node = blockIdx.x * 16 + row;
  int nc = min(node, a.N - 1);

  for (int c = c0; c < c0 + 64; ++c) b0[row][c] = a.msgacc[(size_t)nc * 128 + c];
  gemm_frag(&b0[0][0], a.opf, 8, 0, 4, 0, 8, &b1[0][0], a.opb, false, false, lane); // msg
  for (int c = c0; c < c0 + 64; ++c) b0[row][c] = a.cen[(size_t)nc * 128 + c];      // ce_n
  gemm_frag(&b1[0][0], a.ihf, 8, 0, 4, 0, 8, &b2[0][0], a.ihb, false, false, lane);
  gemm_frag(&b0[0][0], a.hhf, 8, 0, 4, 0, 8, &b2[0][0], a.hhb, true,  false, lane); // gate pre
  gemm_frag(&b0[0][0], a.sf,  8, 0, 4, 0, 8, &b3[0][0], a.sb,  false, false, lane); // self
  __syncthreads();
  for (int c = c0; c < c0 + 64; ++c) {           // msg + sigmoid(gate)*(self-msg)
    float m = b1[row][c];
    float g = 1.f / (1.f + expf(-b2[row][c]));
    b1[row][c] = m + g * (b3[row][c] - m);
  }
  for (int c = c0; c < c0 + 64; ++c)             // ce + msg
    b2[row][c] = a.ce[(size_t)nc * 128 + c] + b1[row][c];
  lnorm(&b2[0][0], a.n2g, a.n2b, false, lane, red);        // h
  for (int c = c0; c < c0 + 64; ++c) b3[row][c] = b2[row][c];  // out = h + ...
  __syncthreads();
  for (int kc = 0; kc < 4; ++kc) {               // MLP: 128->512->128 in 4 chunks
    gemm_frag(&b2[0][0], a.m1f, 32, 0, 4, kc * 8, 8, &b1[0][0],
              a.m1b + kc * 128, false, true, lane);        // relu(l1 chunk)
    gemm_frag(&b1[0][0], a.m2f, 8, kc * 4, 4, 0, 8, &b3[0][0],
              (kc == 0) ? a.m2b : nullptr, true, false, lane);
  }
  if (node < a.N)
    for (int c = c0; c < c0 + 64; ++c) a.out[(size_t)node * 128 + c] = b3[row][c];
}

// ---------------------------------------------------------------------------
// Host launcher
// ---------------------------------------------------------------------------
extern "C" void kernel_launch(void* const* d_in, const int* in_sizes, int n_in,
                              void* d_out, int out_size, void* d_ws, size_t ws_size,
                              hipStream_t stream) {
  (void)n_in; (void)out_size; (void)ws_size;
  const int N = in_sizes[I_NODES] / 2;
  const int E = in_sizes[I_EDGE_INDEX] / 2;

  char* wsb = (char*)d_ws;
  size_t off = 0;
  auto carve = [&](size_t bytes) -> void* {
    size_t o = (off + 255) & ~(size_t)255;
    off = o + bytes;
    return (void*)(wsb + o);
  };

  // f16 fragment weights
  struct WS { int idx; int dout, din; _Float16* dst; } wspec[14] = {
    {I_C_L2_W, 128, 128, nullptr}, {I_C_L3_W, 128, 128, nullptr},
    {I_NN_L2_W, 128, 128, nullptr}, {I_NE_L2_W, 128, 128, nullptr},
    {I_AG_L_W, 128, 128, nullptr},
    {I_Q_W, 128, 128, nullptr}, {I_K_W, 128, 128, nullptr},
    {I_V_W, 128, 128, nullptr}, {I_SELF_W, 128, 128, nullptr},
    {I_IH_W, 128, 128, nullptr}, {I_HH_W, 128, 128, nullptr},
    {I_OUTP_W, 128, 128, nullptr},
    {I_MLP1_W, 512, 128, nullptr}, {I_MLP2_W, 128, 512, nullptr},
  };
  for (int i = 0; i < 14; ++i)
    wspec[i].dst = (_Float16*)carve((size_t)wspec[i].dout * wspec[i].din * sizeof(_Float16));
  float* ce  = (float*)carve((size_t)N * 128 * sizeof(float));
  float* cen = (float*)carve((size_t)N * 128 * sizeof(float));
  float* msg = (float*)carve((size_t)N * 128 * sizeof(float));

  for (int i = 0; i < 14; ++i) {
    int total = wspec[i].dout * wspec[i].din;
    k_prep_weight<<<(total + 255) / 256, 256, 0, stream>>>(
        (const float*)d_in[wspec[i].idx], wspec[i].dst, wspec[i].dout, wspec[i].din);
  }
  k_zero<<<((size_t)N * 128 + 255) / 256, 256, 0, stream>>>(msg, N * 128);

  NodeArgs na;
  na.nodes = (const float*)d_in[I_NODES];
  na.rot   = (const float*)d_in[I_ROT];
  na.l1w = (const float*)d_in[I_C_L1_W]; na.l1b = (const float*)d_in[I_C_L1_B];
  na.ln1g = (const float*)d_in[I_C_LN1_G]; na.ln1b = (const float*)d_in[I_C_LN1_B];
  na.l2f = wspec[0].dst; na.l2b = (const float*)d_in[I_C_L2_B];
  na.ln2g = (const float*)d_in[I_C_LN2_G]; na.ln2b = (const float*)d_in[I_C_LN2_B];
  na.l3f = wspec[1].dst; na.l3b = (const float*)d_in[I_C_L3_B];
  na.ln3g = (const float*)d_in[I_C_LN3_G]; na.ln3b = (const float*)d_in[I_C_LN3_B];
  na.n1g = (const float*)d_in[I_NORM1_G]; na.n1b = (const float*)d_in[I_NORM1_B];
  na.ce = ce; na.cen = cen; na.N = N;
  k_node_embed<<<(N + 15) / 16, 32, 0, stream>>>(na);

  EdgeArgs ea;
  ea.ei = (const int*)d_in[I_EDGE_INDEX];
  ea.nodes = (const float*)d_in[I_NODES];
  ea.eattr = (const float*)d_in[I_EDGE_ATTR];
  ea.rot = (const float*)d_in[I_ROT];
  ea.cen = cen;
  ea.nnl1w = (const float*)d_in[I_NN_L1_W]; ea.nnl1b = (const float*)d_in[I_NN_L1_B];
  ea.nnlng = (const float*)d_in[I_NN_LN_G]; ea.nnlnb = (const float*)d_in[I_NN_LN_B];
  ea.nnl2f = wspec[2].dst; ea.nnl2b = (const float*)d_in[I_NN_L2_B];
  ea.nel1w = (const float*)d_in[I_NE_L1_W]; ea.nel1b = (const float*)d_in[I_NE_L1_B];
  ea.nelng = (const float*)d_in[I_NE_LN_G]; ea.nelnb = (const float*)d_in[I_NE_LN_B];
  ea.nel2f = wspec[3].dst; ea.nel2b = (const float*)d_in[I_NE_L2_B];
  ea.ag1g = (const float*)d_in[I_AG_LN1_G]; ea.ag1b = (const float*)d_in[I_AG_LN1_B];
  ea.agf = wspec[4].dst; ea.agb = (const float*)d_in[I_AG_L_B];
  ea.ag2g = (const float*)d_in[I_AG_LN2_G]; ea.ag2b = (const float*)d_in[I_AG_LN2_B];
  ea.qf = wspec[5].dst; ea.qb = (const float*)d_in[I_Q_B];
  ea.kf = wspec[6].dst; ea.kb = (const float*)d_in[I_K_B];
  ea.vf = wspec[7].dst; ea.vb = (const float*)d_in[I_V_B];
  ea.msg = msg; ea.E = E;
  k_edge<<<(E + 15) / 16, 32, 0, stream>>>(ea);

  OutArgs oa;
  oa.msgacc = msg; oa.ce = ce; oa.cen = cen;
  oa.opf = wspec[11].dst; oa.opb = (const float*)d_in[I_OUTP_B];
  oa.ihf = wspec[9].dst;  oa.ihb = (const float*)d_in[I_IH_B];
  oa.hhf = wspec[10].dst; oa.hhb = (const float*)d_in[I_HH_B];
  oa.sf  = wspec[8].dst;  oa.sb  = (const float*)d_in[I_SELF_B];
  oa.n2g = (const float*)d_in[I_NORM2_G]; oa.n2b = (const float*)d_in[I_NORM2_B];
  oa.m1f = wspec[12].dst; oa.m1b = (const float*)d_in[I_MLP1_B];
  oa.m2f = wspec[13].dst; oa.m2b = (const float*)d_in[I_MLP2_B];
  oa.out = (float*)d_out; oa.N = N;
  k_node_out<<<(N + 15) / 16, 32, 0, stream>>>(oa);
}
